// MSCA_58832462021087
// MI455X (gfx1250) — matrix operand order repaired and checked by reading, not compiled, not run on hardware
//
#include <hip/hip_runtime.h>
#include <hip/hip_bf16.h>

// ---------------------------------------------------------------------------
// MSCA on MI455X (gfx1250): fused implicit-GEMM convs via v_wmma_f32_16x16x32_bf16
// Dims fixed by the reference: B=8, C=256, H=W=128, G=32 -> N=256 images, cg=8
// ---------------------------------------------------------------------------

#define NIMG   256      // B * groups
#define CG     8
#define NC     2048     // NIMG * CG
#define HW     16384
#define HDIM   128
#define WDIM   128

typedef __attribute__((ext_vector_type(16))) __bf16 v16b;
typedef __attribute__((ext_vector_type(8)))  float  v8f;

__device__ __forceinline__ float msca_sigmoid(float x) {
    return 1.0f / (1.0f + __expf(-x));
}
__device__ __forceinline__ float msca_wave_sum(float v) {
    #pragma unroll
    for (int o = 16; o > 0; o >>= 1) v += __shfl_xor(v, o, 32);
    return v;
}

// ---------------------------------------------------------------------------
// P0: zero the cross-launch accumulators (s1,s2,t1,t2 = 4*2048 floats)
// ---------------------------------------------------------------------------
__global__ void msca_zero(float* __restrict__ a, int len) {
    int i = blockIdx.x * 256 + threadIdx.x;
    if (i < len) a[i] = 0.0f;
}

// ---------------------------------------------------------------------------
// P1: per-(n,c) row means (x_h) and column means (x_w), float4 global loads.
// diag_pool is provably dead (fuse conv is 1x1; rows >= H+W are discarded).
// ---------------------------------------------------------------------------
__global__ void msca_pool(const float* __restrict__ x, float* __restrict__ pool) {
    __shared__ float lh[HDIM], lw[WDIM];
    const int nc   = blockIdx.x;
    const int t    = threadIdx.x;
    const int lane = t & 31;
    if (t < 128) lw[t] = 0.0f;
    __syncthreads();
    const float4* px4 = (const float4*)(x + (size_t)nc * HW);
    float c0 = 0.f, c1 = 0.f, c2 = 0.f, c3 = 0.f;
    #pragma unroll 4
    for (int i = 0; i < 16; ++i) {
        int q = t + (i << 8);                 // float4 index; pixel p = 4q
        float4 v = px4[q];
        c0 += v.x; c1 += v.y; c2 += v.z; c3 += v.w;
        float rv = msca_wave_sum(v.x + v.y + v.z + v.w);
        if (lane == 0) lh[q >> 5] = rv;       // row = p>>7 = q>>5 (one wave per row)
    }
    const int cb = lane << 2;                 // 4 owned columns, constant per thread
    atomicAdd(&lw[cb + 0], c0);
    atomicAdd(&lw[cb + 1], c1);
    atomicAdd(&lw[cb + 2], c2);
    atomicAdd(&lw[cb + 3], c3);
    __syncthreads();
    if (t < 128) {
        pool[(size_t)nc * 256 + t]       = lh[t] * (1.0f / 128.0f);
        pool[(size_t)nc * 256 + 128 + t] = lw[t] * (1.0f / 128.0f);
    }
}

// ---------------------------------------------------------------------------
// P2: axis-fuse 1x1 conv (8x8 channel mix) + sigmoid -> gates g_h, g_w
// ---------------------------------------------------------------------------
__global__ void msca_fuse(const float* __restrict__ pool,
                          const float* __restrict__ afw,
                          float* __restrict__ gate) {
    const int n = blockIdx.x;
    const int t = threadIdx.x;                 // 0..255 = fused row (h or 128+w)
    float pin[CG];
    #pragma unroll
    for (int ci = 0; ci < CG; ++ci) pin[ci] = pool[((size_t)(n * CG + ci)) * 256 + t];
    #pragma unroll
    for (int co = 0; co < CG; ++co) {
        float o = 0.0f;
        #pragma unroll
        for (int ci = 0; ci < CG; ++ci) o += afw[co * CG + ci] * pin[ci];
        gate[((size_t)(n * CG + co)) * 256 + t] = msca_sigmoid(o);
    }
}

// ---------------------------------------------------------------------------
// P3: group-norm stats of (gx * g_h * g_w) per (n,c), float4 global loads
// ---------------------------------------------------------------------------
__global__ void msca_stats(const float* __restrict__ x,
                           const float* __restrict__ gate,
                           float* __restrict__ mu, float* __restrict__ rs) {
    const int nc   = blockIdx.x;
    const int t    = threadIdx.x;
    const int lane = t & 31;
    const float4* px4 = (const float4*)(x + (size_t)nc * HW);
    const float*  gh  = gate + (size_t)nc * 256;
    const float4  gwv = *(const float4*)&gate[(size_t)nc * 256 + 128 + (lane << 2)];
    float s = 0.0f, q = 0.0f;
    #pragma unroll 4
    for (int i = 0; i < 16; ++i) {
        int qi = t + (i << 8);
        float4 v4  = px4[qi];
        float  ghv = gh[qi >> 5];
        float a = v4.x * ghv * gwv.x;
        float b = v4.y * ghv * gwv.y;
        float c = v4.z * ghv * gwv.z;
        float d = v4.w * ghv * gwv.w;
        s += a + b + c + d;
        q += a * a + b * b + c * c + d * d;
    }
    s = msca_wave_sum(s);
    q = msca_wave_sum(q);
    __shared__ float ls[8], lq[8];
    if (lane == 0) { ls[t >> 5] = s; lq[t >> 5] = q; }
    __syncthreads();
    if (t == 0) {
        float S = 0.0f, Q = 0.0f;
        #pragma unroll
        for (int i = 0; i < 8; ++i) { S += ls[i]; Q += lq[i]; }
        float m   = S * (1.0f / 16384.0f);
        float var = Q * (1.0f / 16384.0f) - m * m;
        mu[nc] = m;
        rs[nc] = rsqrtf(var + 1e-5f);
    }
}

// ---------------------------------------------------------------------------
// P4: the flagship kernel. Implicit GEMM via WMMA:
//   combined B[K=224][16]: cols 0..7 = conv3x3 embedded in 5x5 footprint,
//                          cols 8..15 = conv5x5  -> 7 x v_wmma K=32 steps
//   plus one WMMA for conv1x1 on the normalized gated tensor.
// A-fragment LDS offsets constant-fold per (s,i,khalf); B fragments are
// single aligned v16b LDS loads from a column-major staging buffer.
// ---------------------------------------------------------------------------
#define TSTR  40                 // padded LDS row stride (bf16) for 36-wide halo tile
#define CHSTR (36 * TSTR)        // 1440
#define NSTR  36                 // normalized tile row stride
#define NCHS  (32 * NSTR)        // 1152

// (ci,ky,kx) -> LDS offset inside the halo tile; k >= 200 is padding (B rows zero)
constexpr int msca_koff(int k) {
    return (k >= 200) ? 0
        : ((k / 25) * CHSTR + (((k % 25) / 5) * TSTR) + ((k % 25) % 5));
}

__launch_bounds__(256)
__global__ void msca_conv_wmma(const float* __restrict__ x,
                               const float* __restrict__ gate,
                               const float* __restrict__ mu,
                               const float* __restrict__ rs,
                               const float* __restrict__ gnw,
                               const float* __restrict__ gnb,
                               const float* __restrict__ w1,
                               const float* __restrict__ w3,
                               const float* __restrict__ w5,
                               const float* __restrict__ scw,
                               float* __restrict__ x1p,
                               float* __restrict__ x2p,
                               float* __restrict__ s1g,
                               float* __restrict__ s2g) {
    __shared__ __bf16 tile[CG * 36 * TSTR];          // 8ch x 36x36 halo tile (bf16)
    __shared__ __bf16 ntile[CG * 32 * NSTR];         // normalized gated tile (conv1x1)
    alignas(32) __shared__ __bf16 BshT[16 * 224];    // combined weights, [col][k]
    alignas(32) __shared__ __bf16 BashT[16 * 32];    // conv1x1 weights,  [col][k]
    __shared__ float  s1l[CG], s2l[CG];

    const int tid = threadIdx.x;
    const int n   = blockIdx.x >> 4;
    const int tl  = blockIdx.x & 15;
    const int ty0 = (tl >> 2) << 5;
    const int tx0 = (tl & 3) << 5;

    // softmax(scale_w) (3 elements, uniform across block)
    float e0 = __expf(scw[0]), e1 = __expf(scw[1]), e2 = __expf(scw[2]);
    float ei = 1.0f / (e0 + e1 + e2);
    const float sw0 = e0 * ei, sw1 = e1 * ei, sw2 = e2 * ei;

    if (tid < CG) { s1l[tid] = 0.0f; s2l[tid] = 0.0f; }

    // ---- stage halo tile (f32 -> bf16) -------------------------------------
    for (int idx = tid; idx < CG * 36 * 36; idx += 256) {
        int ci = idx / 1296;
        int r  = idx - ci * 1296;
        int yy = r / 36;
        int xx = r - yy * 36;
        int gy = ty0 + yy - 2;
        int gx_ = tx0 + xx - 2;
        float v = 0.0f;
        if (gy >= 0 && gy < HDIM && gx_ >= 0 && gx_ < WDIM) {
            const float* sp = &x[((size_t)(n * CG + ci) * HDIM + gy) * WDIM + gx_];
            __builtin_prefetch(sp + WDIM, 0, 1);   // global_prefetch_b8 next row
            v = *sp;
        }
        tile[ci * CHSTR + yy * TSTR + xx] = (__bf16)v;
    }
    // ---- combined B, column-major [j][k]: j<8 -> 3x3 embedded, j>=8 -> 5x5 -
    for (int idx = tid; idx < 16 * 224; idx += 256) {
        int j = idx / 224, k = idx - j * 224;
        float v = 0.0f;
        if (k < 200) {
            int ci = k / 25, r = k - ci * 25, ky = r / 5, kx = r - ky * 5;
            if (j < 8) {
                if (ky >= 1 && ky <= 3 && kx >= 1 && kx <= 3)
                    v = w3[((j * CG + ci) * 3 + (ky - 1)) * 3 + (kx - 1)];
            } else {
                v = w5[(((j - 8) * CG + ci) * 5 + ky) * 5 + kx];
            }
        }
        BshT[idx] = (__bf16)v;
    }
    for (int idx = tid; idx < 16 * 32; idx += 256) {
        int j = idx >> 5, k = idx & 31;
        float v = (k < 8 && j < 8) ? w1[j * CG + k] : 0.0f;
        BashT[idx] = (__bf16)v;
    }
    __syncthreads();

    // ---- normalized gated tile for conv1x1 (reads staged tile) -------------
    {
        const int ncb = n * CG;
        for (int idx = tid; idx < CG * 32 * 32; idx += 256) {
            int ci = idx >> 10;
            int r  = idx & 1023;
            int yy = r >> 5, xx = r & 31;
            float v  = (float)tile[ci * CHSTR + (yy + 2) * TSTR + (xx + 2)];
            float gh = gate[(size_t)(ncb + ci) * 256 + (ty0 + yy)];
            float gw = gate[(size_t)(ncb + ci) * 256 + 128 + (tx0 + xx)];
            float vg = v * gh * gw;
            float nv = (vg - mu[ncb + ci]) * rs[ncb + ci] * gnw[ci] + gnb[ci];
            ntile[ci * NCHS + yy * NSTR + xx] = (__bf16)nv;
        }
    }
    __syncthreads();

    const int lane  = tid & 31;
    const int wid   = tid >> 5;
    const int m     = lane & 15;     // A row (pixel within 16-pixel M-tile)
    const int khalf = lane >> 4;     // A: K half; B: lanes 0-15 K0..15, 16-31 K16..31
    const int ncol  = lane & 15;     // B/C/D column

    // ---- B fragments: one aligned v16b LDS load each (2x ds_load_b128) -----
    const v16b* bAll  = (const v16b*)BshT;    // col stride 224 bf16 = 14 v16b
    const v16b* baAll = (const v16b*)BashT;   // col stride  32 bf16 =  2 v16b
    v16b Bc[7], Baf;
    #pragma unroll
    for (int s = 0; s < 7; ++s)
        Bc[s] = bAll[ncol * 14 + (s << 1) + khalf];
    Baf = baAll[(ncol << 1) + khalf];

    // ---- 8 M-tiles per wave: 16 consecutive pixels each --------------------
    #pragma unroll 1
    for (int t8 = 0; t8 < 8; ++t8) {
        const int py   = (wid << 2) + (t8 >> 1);
        const int px0  = (t8 & 1) << 4;
        const int pixB = py * TSTR + px0 + m;
        const int npix = py * NSTR + px0 + m;

        v8f acc = {0.f, 0.f, 0.f, 0.f, 0.f, 0.f, 0.f, 0.f};
        #pragma unroll
        for (int s = 0; s < 7; ++s) {
            v16b Af;
            #pragma unroll
            for (int i = 0; i < 16; ++i) {
                // A layout: lane m holds K {0..15}; lane m+16 holds {16..31}
                const int kl0 = i;
                const int o0  = msca_koff((s << 5) + kl0);       // khalf = 0
                const int o1  = msca_koff((s << 5) + kl0 + 16);  // khalf = 1
                Af[i] = tile[(khalf ? o1 : o0) + pixB];
            }
            acc = __builtin_amdgcn_wmma_f32_16x16x32_bf16(
                      false, Af, false, Bc[s], (short)0, acc, false, false);
        }
        // conv1x1 on normalized tile (real K=0..7 live only in lanes 0..15)
        v16b Aa;
        #pragma unroll
        for (int i = 0; i < 16; ++i) {
            __bf16 z = (__bf16)0.0f;
            Aa[i] = (i < 8 && khalf == 0) ? ntile[i * NCHS + npix] : z;
        }
        v8f acca = {0.f, 0.f, 0.f, 0.f, 0.f, 0.f, 0.f, 0.f};
        acca = __builtin_amdgcn_wmma_f32_16x16x32_bf16(
                   false, Aa, false, Baf, (short)0, acca, false, false);

        // cols 0..7 = xb (3x3), cols 8..15 = xc (5x5); pull xc across with xor-8
        float x1v[8], x2v[8];
        #pragma unroll
        for (int r = 0; r < 8; ++r) {
            float xb = acc[r];
            float xc = __shfl_xor(xb, 8, 32);
            x1v[r] = sw0 * acca[r] + sw1 * xb + sw2 * xc;
            x2v[r] = xb;
        }
        if ((lane & 8) == 0) {                    // writer lanes own channels 0..7
            int c   = lane & 7;
            int y   = ty0 + py;
            int xb0 = tx0 + px0 + (khalf << 3);   // C rows M = r + 8*khalf
            size_t bp = ((size_t)(n * CG + c) * HDIM + y) * WDIM + xb0;
            float4 a0 = make_float4(x1v[0], x1v[1], x1v[2], x1v[3]);
            float4 a1 = make_float4(x1v[4], x1v[5], x1v[6], x1v[7]);
            float4 b0 = make_float4(x2v[0], x2v[1], x2v[2], x2v[3]);
            float4 b1 = make_float4(x2v[4], x2v[5], x2v[6], x2v[7]);
            *(float4*)(x1p + bp)     = a0;
            *(float4*)(x1p + bp + 4) = a1;
            *(float4*)(x2p + bp)     = b0;
            *(float4*)(x2p + bp + 4) = b1;
            float l1 = (a0.x + a0.y + a0.z + a0.w) + (a1.x + a1.y + a1.z + a1.w);
            float l2 = (b0.x + b0.y + b0.z + b0.w) + (b1.x + b1.y + b1.z + b1.w);
            atomicAdd(&s1l[c], l1);               // ds_add_f32
            atomicAdd(&s2l[c], l2);
        }
    }
    __syncthreads();
    if (tid < CG) {
        atomicAdd(&s1g[n * CG + tid], s1l[tid]);
        atomicAdd(&s2g[n * CG + tid], s2l[tid]);
    }
}

// ---------------------------------------------------------------------------
// P5: channel gates from HW-means (8x8 mix + sigmoid), shared weights
// ---------------------------------------------------------------------------
__global__ void msca_chgate(const float* __restrict__ s1g, const float* __restrict__ s2g,
                            const float* __restrict__ wch,
                            float* __restrict__ chg1, float* __restrict__ chg2) {
    const int n = blockIdx.x;
    const int t = threadIdx.x;
    if (t < CG) {
        float a = 0.0f, b = 0.0f;
        #pragma unroll
        for (int ci = 0; ci < CG; ++ci) {
            float w = wch[t * CG + ci];
            a += w * (s1g[n * CG + ci] * (1.0f / 16384.0f));
            b += w * (s2g[n * CG + ci] * (1.0f / 16384.0f));
        }
        chg1[n * CG + t] = msca_sigmoid(a);
        chg2[n * CG + t] = msca_sigmoid(b);
    }
}

// ---------------------------------------------------------------------------
// P6: apply gates, accumulate gated HW-means (for the 8-wide softmax)
// ---------------------------------------------------------------------------
__global__ void msca_gatedsum(const float* __restrict__ x1p, const float* __restrict__ x2p,
                              const float* __restrict__ chg1, const float* __restrict__ chg2,
                              const float* __restrict__ wsp,
                              float* __restrict__ t1g, float* __restrict__ t2g) {
    const int n = blockIdx.x >> 6;
    const int p = ((blockIdx.x & 63) << 8) + threadIdx.x;
    const float* b1 = x1p + (size_t)n * CG * HW;
    const float* b2 = x2p + (size_t)n * CG * HW;
    float v1[CG], v2[CG], d1 = 0.0f, d2 = 0.0f;
    #pragma unroll
    for (int c = 0; c < CG; ++c) {
        v1[c] = b1[(size_t)c * HW + p];
        v2[c] = b2[(size_t)c * HW + p];
        d1 += wsp[c] * v1[c];
        d2 += wsp[c] * v2[c];
    }
    float sp1 = msca_sigmoid(d1), sp2 = msca_sigmoid(d2);
    __shared__ float l1[CG], l2[CG];
    if (threadIdx.x < CG) { l1[threadIdx.x] = 0.0f; l2[threadIdx.x] = 0.0f; }
    __syncthreads();
    const int lane = threadIdx.x & 31;
    #pragma unroll
    for (int c = 0; c < CG; ++c) {
        float g1 = msca_wave_sum(v1[c] * chg1[n * CG + c] * sp1);
        float g2 = msca_wave_sum(v2[c] * chg2[n * CG + c] * sp2);
        if (lane == 0) { atomicAdd(&l1[c], g1); atomicAdd(&l2[c], g2); }
    }
    __syncthreads();
    if (threadIdx.x < CG) {
        atomicAdd(&t1g[n * CG + threadIdx.x], l1[threadIdx.x]);
        atomicAdd(&t2g[n * CG + threadIdx.x], l2[threadIdx.x]);
    }
}

// ---------------------------------------------------------------------------
// P7: 8-wide softmax of gated means
// ---------------------------------------------------------------------------
__global__ void msca_softmax8(const float* __restrict__ t1g, const float* __restrict__ t2g,
                              float* __restrict__ a1, float* __restrict__ a2) {
    const int n = blockIdx.x;
    if (threadIdx.x == 0) {
        float u1[CG], u2[CG], m1 = -1e30f, m2 = -1e30f;
        #pragma unroll
        for (int c = 0; c < CG; ++c) {
            u1[c] = t1g[n * CG + c] * (1.0f / 16384.0f);
            u2[c] = t2g[n * CG + c] * (1.0f / 16384.0f);
            m1 = fmaxf(m1, u1[c]); m2 = fmaxf(m2, u2[c]);
        }
        float s1 = 0.0f, s2 = 0.0f;
        #pragma unroll
        for (int c = 0; c < CG; ++c) {
            u1[c] = __expf(u1[c] - m1); u2[c] = __expf(u2[c] - m2);
            s1 += u1[c]; s2 += u2[c];
        }
        #pragma unroll
        for (int c = 0; c < CG; ++c) {
            a1[n * CG + c] = u1[c] / s1;
            a2[n * CG + c] = u2[c] / s2;
        }
    }
}

// ---------------------------------------------------------------------------
// P8: weights = a1 . x2g + a2 . x1g per pixel; out = gx * sigmoid(weights)
// (gates recomputed from scalars to avoid an extra 268MB of HBM writes)
// ---------------------------------------------------------------------------
__global__ void msca_final(const float* __restrict__ x,
                           const float* __restrict__ x1p, const float* __restrict__ x2p,
                           const float* __restrict__ chg1, const float* __restrict__ chg2,
                           const float* __restrict__ wsp,
                           const float* __restrict__ a1, const float* __restrict__ a2,
                           float* __restrict__ out) {
    const int n = blockIdx.x >> 6;
    const int p = ((blockIdx.x & 63) << 8) + threadIdx.x;
    const float* b1 = x1p + (size_t)n * CG * HW;
    const float* b2 = x2p + (size_t)n * CG * HW;
    const float* bx = x   + (size_t)n * CG * HW;
    float v1[CG], v2[CG], gxv[CG], d1 = 0.0f, d2 = 0.0f;
    #pragma unroll
    for (int c = 0; c < CG; ++c) {
        v1[c]  = b1[(size_t)c * HW + p];
        v2[c]  = b2[(size_t)c * HW + p];
        gxv[c] = bx[(size_t)c * HW + p];
        d1 += wsp[c] * v1[c];
        d2 += wsp[c] * v2[c];
    }
    float sp1 = msca_sigmoid(d1), sp2 = msca_sigmoid(d2);
    float wsum = 0.0f;
    #pragma unroll
    for (int c = 0; c < CG; ++c) {
        wsum += a1[n * CG + c] * (v2[c] * chg2[n * CG + c] * sp2)
              + a2[n * CG + c] * (v1[c] * chg1[n * CG + c] * sp1);
    }
    float sg = msca_sigmoid(wsum);
    #pragma unroll
    for (int c = 0; c < CG; ++c)
        out[(size_t)(n * CG + c) * HW + p] = gxv[c] * sg;
}

// ---------------------------------------------------------------------------
// Workspace layout (floats)
// ---------------------------------------------------------------------------
#define OFF_POOL  ((size_t)0)
#define OFF_GATE  (OFF_POOL + (size_t)NC * 256)
#define OFF_MU    (OFF_GATE + (size_t)NC * 256)
#define OFF_RS    (OFF_MU   + NC)
#define OFF_S1    (OFF_RS   + NC)
#define OFF_S2    (OFF_S1   + NC)
#define OFF_T1    (OFF_S2   + NC)
#define OFF_T2    (OFF_T1   + NC)
#define OFF_CHG1  (OFF_T2   + NC)
#define OFF_CHG2  (OFF_CHG1 + NC)
#define OFF_A1    (OFF_CHG2 + NC)
#define OFF_A2    (OFF_A1   + NC)
#define OFF_X1    (OFF_A2   + NC)
#define OFF_X2    (OFF_X1   + (size_t)NC * HW)

extern "C" void kernel_launch(void* const* d_in, const int* in_sizes, int n_in,
                              void* d_out, int out_size, void* d_ws, size_t ws_size,
                              hipStream_t stream) {
    (void)in_sizes; (void)n_in; (void)out_size; (void)ws_size;
    const float* x   = (const float*)d_in[0];
    const float* afw = (const float*)d_in[1];
    const float* gnw = (const float*)d_in[2];
    const float* gnb = (const float*)d_in[3];
    const float* w1  = (const float*)d_in[4];
    const float* w3  = (const float*)d_in[5];
    const float* w5  = (const float*)d_in[6];
    const float* scw = (const float*)d_in[7];
    const float* wch = (const float*)d_in[8];
    const float* wsp = (const float*)d_in[9];
    float* out = (float*)d_out;

    float* ws   = (float*)d_ws;
    float* pool = ws + OFF_POOL;
    float* gate = ws + OFF_GATE;
    float* mu   = ws + OFF_MU;
    float* rs   = ws + OFF_RS;
    float* s1   = ws + OFF_S1;
    float* s2   = ws + OFF_S2;
    float* t1   = ws + OFF_T1;
    float* t2   = ws + OFF_T2;
    float* chg1 = ws + OFF_CHG1;
    float* chg2 = ws + OFF_CHG2;
    float* a1   = ws + OFF_A1;
    float* a2   = ws + OFF_A2;
    float* x1p  = ws + OFF_X1;
    float* x2p  = ws + OFF_X2;

    // s1,s2,t1,t2 are contiguous: zero all 4 * NC floats every launch
    msca_zero<<<(4 * NC + 255) / 256, 256, 0, stream>>>(s1, 4 * NC);
    msca_pool<<<NC, 256, 0, stream>>>(x, pool);
    msca_fuse<<<NIMG, 256, 0, stream>>>(pool, afw, gate);
    msca_stats<<<NC, 256, 0, stream>>>(x, gate, mu, rs);
    msca_conv_wmma<<<NIMG * 16, 256, 0, stream>>>(x, gate, mu, rs, gnw, gnb,
                                                  w1, w3, w5, scw,
                                                  x1p, x2p, s1, s2);
    msca_chgate<<<NIMG, 32, 0, stream>>>(s1, s2, wch, chg1, chg2);
    msca_gatedsum<<<NIMG * 64, 256, 0, stream>>>(x1p, x2p, chg1, chg2, wsp, t1, t2);
    msca_softmax8<<<NIMG, 32, 0, stream>>>(t1, t2, a1, a2);
    msca_final<<<NIMG * 64, 256, 0, stream>>>(x, x1p, x2p, chg1, chg2, wsp, a1, a2, out);
}
